// MPMC_net_47888885351048
// MI455X (gfx1250) — matrix operand order repaired and checked by reading, not compiled, not run on hardware
//
#include <hip/hip_runtime.h>

#define NN 32768      // nodes
#define NE 1048576    // edges
#define NH 64         // hidden
#define EPSN 1e-5f

typedef _Float16 f16;
typedef unsigned int u32;
typedef __attribute__((ext_vector_type(16))) _Float16 v16h;
typedef __attribute__((ext_vector_type(8)))  _Float16 v8h;
typedef __attribute__((ext_vector_type(8)))  float    v8f;
typedef __attribute__((ext_vector_type(4)))  u32      u32x4;
typedef __attribute__((ext_vector_type(8)))  int      i32x8;
typedef __attribute__((ext_vector_type(4)))  int      i32x4;

static __device__ __forceinline__ v8f wmma_f16(v16h a, v16h b, v8f c) {
  return __builtin_amdgcn_wmma_f32_16x16x32_f16(false, a, false, b, (short)0, c,
                                                false, false);
}

// low 32 bits of a generic shared pointer == AS3 byte offset within the
// workgroup LDS allocation (flat-shared = aperture<<32 | lds_offset)
static __device__ __forceinline__ u32 lds_off(const void* p) {
  return (u32)(unsigned long long)(uintptr_t)p;
}

// ---------------------------------------------------------------------------
// TDM: DMA `ndw` dwords from global memory into LDS at byte offset `laddr`.
// D# per CDNA5 ISA Sec 8.3/8.4: 1-row 2D tile, data_size=4B, no pad/iterate,
// groups 2/3 zero (<=2D tensor). Issued once by one wave; completion via
// s_wait_tensorcnt. (6-arg builtin signature: clang-23 / therock headers.)
// ---------------------------------------------------------------------------
static __device__ __forceinline__ void tdm_load_1d(const void* gptr, u32 laddr,
                                                   u32 ndw) {
  unsigned long long ga = (unsigned long long)(uintptr_t)gptr;
  u32x4 g0;
  g0[0] = 1u;                                            // count=1 (valid user D#)
  g0[1] = laddr;                                         // lds_addr
  g0[2] = (u32)(ga & 0xffffffffu);                       // global_addr[31:0]
  g0[3] = (u32)((ga >> 32) & 0x1ffffffu) | (2u << 30);   // global_addr[56:32], type=2
  i32x8 g1;
  g1[0] = (int)(2u << 16);                 // data_size = 4 bytes
  g1[1] = (int)((ndw & 0xffffu) << 16);    // tensor_dim0[15:0]
  g1[2] = (int)((ndw >> 16) & 0xffffu)     // tensor_dim0[31:16]
        | (int)(1u << 16);                 // tensor_dim1 = 1
  g1[3] = (int)((ndw & 0xffffu) << 16);    // tile_dim0 = ndw
  g1[4] = 1;                               // tile_dim1 = 1, tile_dim2 = 0
  g1[5] = (int)ndw;                        // tensor_dim0_stride[31:0]
  g1[6] = 0;
  g1[7] = 0;
  i32x4 z4 = {0, 0, 0, 0};
  i32x8 z8 = {0, 0, 0, 0, 0, 0, 0, 0};
  __builtin_amdgcn_tensor_load_to_lds(g0, g1, z4, z4, z8, 0);
}

// ---------------------------------------------------------------------------
// Pack an f32 weight W[K][Nw] (row-major) into WMMA B-fragment order:
// frag = kt*(Nw/16)+nt, each frag = 32 lanes x 16 halfs.
// lane l, elem i: K = kt*32 + (i & 7) + (l>=16 ? 8 : 0) + (i>=8 ? 16 : 0), N = nt*16 + (l & 15)
// ---------------------------------------------------------------------------
__global__ void pack_b_kernel(const float* __restrict__ W, f16* __restrict__ dst,
                              int K, int Nw) {
  int tid = blockIdx.x * blockDim.x + threadIdx.x;
  if (tid >= K * Nw) return;
  int ntiles = Nw >> 4;
  int frag   = tid >> 9;
  int within = tid & 511;
  int lane   = within >> 4;
  int i      = within & 15;
  int kt = frag / ntiles;
  int nt = frag - kt * ntiles;
  int k = kt * 32 + (i & 7) + ((lane >= 16) ? 8 : 0) + ((i >= 8) ? 16 : 0);
  int n = nt * 16 + (lane & 15);
  dst[tid] = (f16)W[k * Nw + n];
}

// ---------------------------------------------------------------------------
// Encoder: h = x @ enc_W + enc_b   (K=8 -> scalar f32)
// ---------------------------------------------------------------------------
__global__ void enc_kernel(const float* __restrict__ x, const float* __restrict__ W,
                           const float* __restrict__ b, float* __restrict__ hf,
                           f16* __restrict__ hh) {
  int idx = blockIdx.x * blockDim.x + threadIdx.x;   // node*64 + f
  int f = idx & 63, node = idx >> 6;
  float acc = b[f];
#pragma unroll
  for (int k = 0; k < 8; ++k) acc += x[node * 8 + k] * W[k * 64 + f];
  hf[idx] = acc;
  hh[idx] = (f16)acc;
}

// ---------------------------------------------------------------------------
// Zero agg + per-graph moment buffers
// ---------------------------------------------------------------------------
__global__ void zero_kernel(float* __restrict__ agg, float* __restrict__ gsum,
                            float* __restrict__ gsq) {
  int idx = blockIdx.x * blockDim.x + threadIdx.x;
  if (idx < NN * NH) agg[idx] = 0.f;
  if (idx < 8 * NH) { gsum[idx] = 0.f; gsq[idx] = 0.f; }
}

// ---------------------------------------------------------------------------
// Edge message kernel: per 16-edge tile (one wave):
//   m = relu(relu(cat(h[dst],h[src]) @ W1 + b1) @ W2 + b2); atomic agg[dst]+=m
// Weights staged block-wide into LDS via TDM (24 KB), B-frags read via ds_load.
// GEMM1: M=16 edges, K=128 (4 frags), N=64 (4 frags)  -> 16 WMMA
// GEMM2: K=64 (2 frags), N=64                          ->  8 WMMA
// ---------------------------------------------------------------------------
__global__ __launch_bounds__(256) void edge_msg_kernel(
    const f16* __restrict__ hh, const int* __restrict__ ei,
    const f16* __restrict__ w1, const f16* __restrict__ w2,
    const float* __restrict__ b1, const float* __restrict__ b2,
    float* __restrict__ agg) {
  __shared__ alignas(32) f16 sw1[128 * NH];      // 16 KB packed msg1 frags
  __shared__ alignas(32) f16 sw2[NH * NH];       //  8 KB packed msg2 frags
  __shared__ alignas(32) f16 smid[8][16 * NH];   // 16 KB GEMM1->GEMM2 transpose
  __shared__ int sdst[8][16];

  const int lane = threadIdx.x & 31;
  const int wv   = threadIdx.x >> 5;
  const int m    = lane & 15;     // row (edge) within tile
  const int hi   = lane >> 4;     // half-wave selector
  const int tile = blockIdx.x * 8 + wv;
  const int e    = tile * 16 + m;

  // one wave DMAs both weight blocks into LDS (TDM, TENSORcnt per-wave)
  if (wv == 0) {
    tdm_load_1d(w1, lds_off(sw1), 128 * NH * 2 / 4);
    tdm_load_1d(w2, lds_off(sw2), NH * NH * 2 / 4);
    __builtin_amdgcn_s_wait_tensorcnt(0);
  }

  const int sN = ei[e];           // x_j = h[src]
  const int dN = ei[NE + e];      // x_i = h[dst]
  if (hi == 0) sdst[wv][m] = dN;
  __syncthreads();                // publish LDS weights + sdst

  const f16* drow = hh + dN * NH;
  const f16* srow = hh + sN * NH;

  // ---- GEMM1: cat(h[dst], h[src]) [16x128] @ W1 [128x64] ----
  v8f acc[4] = {};
#pragma unroll
  for (int k = 0; k < 4; ++k) {
    const f16* row = (k < 2) ? drow : srow;
    const int base = (k & 1) * 32;
    v8h c0 = *(const v8h*)(row + base + hi * 8);
    v8h c1 = *(const v8h*)(row + base + 16 + hi * 8);
    v16h a;
#pragma unroll
    for (int i = 0; i < 8; ++i) { a[i] = c0[i]; a[i + 8] = c1[i]; }
#pragma unroll
    for (int nt = 0; nt < 4; ++nt) {
      v16h b = *(const v16h*)(sw1 + (k * 4 + nt) * 512 + lane * 16);
      acc[nt] = wmma_f16(a, b, acc[nt]);
    }
  }
  // bias + relu -> LDS (transpose from C layout to row-major)
#pragma unroll
  for (int nt = 0; nt < 4; ++nt) {
    const int n = nt * 16 + m;
    const float bias = b1[n];
#pragma unroll
    for (int r = 0; r < 8; ++r) {
      float v = acc[nt][r] + bias;
      v = v > 0.f ? v : 0.f;
      smid[wv][(r + hi * 8) * NH + n] = (f16)v;
    }
  }
  __syncthreads();

  // ---- GEMM2: mid [16x64] @ W2 [64x64] ----
  v8f acc2[4] = {};
#pragma unroll
  for (int k = 0; k < 2; ++k) {
    v8h c0 = *(const v8h*)(&smid[wv][m * NH + k * 32 + hi * 8]);
    v8h c1 = *(const v8h*)(&smid[wv][m * NH + k * 32 + 16 + hi * 8]);
    v16h a;
#pragma unroll
    for (int i = 0; i < 8; ++i) { a[i] = c0[i]; a[i + 8] = c1[i]; }
#pragma unroll
    for (int nt = 0; nt < 4; ++nt) {
      v16h b = *(const v16h*)(sw2 + (k * 4 + nt) * 512 + lane * 16);
      acc2[nt] = wmma_f16(a, b, acc2[nt]);
    }
  }
  // bias + relu + scatter-add into agg[dst] (hw global_atomic_add_f32)
#pragma unroll
  for (int nt = 0; nt < 4; ++nt) {
    const int n = nt * 16 + m;
    const float bias = b2[n];
#pragma unroll
    for (int r = 0; r < 8; ++r) {
      float v = acc2[nt][r] + bias;
      v = v > 0.f ? v : 0.f;
      const int dd = sdst[wv][r + hi * 8];
      unsafeAtomicAdd(&agg[dd * NH + n], v);
    }
  }
}

// ---------------------------------------------------------------------------
// Node update kernel: u = relu(relu(cat(h,agg) @ U1 + b1) @ U2 + b2)
// + per-graph sum / sum-of-squares reduction for InstanceNorm
// ---------------------------------------------------------------------------
__global__ __launch_bounds__(256) void node_upd_kernel(
    const f16* __restrict__ hh, const float* __restrict__ agg,
    const f16* __restrict__ w1, const f16* __restrict__ w2,
    const float* __restrict__ b1, const float* __restrict__ b2,
    float* __restrict__ u, float* __restrict__ gsum, float* __restrict__ gsq) {
  __shared__ alignas(32) f16 sw1[128 * NH];
  __shared__ alignas(32) f16 sw2[NH * NH];
  __shared__ alignas(32) f16 smid[8][16 * NH];

  const int lane = threadIdx.x & 31;
  const int wv   = threadIdx.x >> 5;
  const int m    = lane & 15;
  const int hi   = lane >> 4;
  const int tile = blockIdx.x * 8 + wv;
  const int node = tile * 16 + m;
  const int g    = tile >> 8;    // 4096 nodes/graph, 256 tiles/graph

  if (wv == 0) {
    tdm_load_1d(w1, lds_off(sw1), 128 * NH * 2 / 4);
    tdm_load_1d(w2, lds_off(sw2), NH * NH * 2 / 4);
    __builtin_amdgcn_s_wait_tensorcnt(0);
  }
  __syncthreads();

  const f16*   hrow = hh + node * NH;
  const float* arow = agg + node * NH;

  // ---- GEMM1: cat(h, agg) [16x128] @ U1 [128x64] ----
  v8f acc[4] = {};
#pragma unroll
  for (int k = 0; k < 4; ++k) {
    v16h a;
    if (k < 2) {
      const int base = k * 32;
      v8h c0 = *(const v8h*)(hrow + base + hi * 8);
      v8h c1 = *(const v8h*)(hrow + base + 16 + hi * 8);
#pragma unroll
      for (int i = 0; i < 8; ++i) { a[i] = c0[i]; a[i + 8] = c1[i]; }
    } else {
      const int base = (k - 2) * 32;
      const float* p = arow + base + hi * 8;
      const float* q = arow + base + 16 + hi * 8;
      float4 f0 = *(const float4*)(p);
      float4 f1 = *(const float4*)(p + 4);
      float4 g0 = *(const float4*)(q);
      float4 g1 = *(const float4*)(q + 4);
      a[0] = (f16)f0.x; a[1] = (f16)f0.y; a[2] = (f16)f0.z; a[3] = (f16)f0.w;
      a[4] = (f16)f1.x; a[5] = (f16)f1.y; a[6] = (f16)f1.z; a[7] = (f16)f1.w;
      a[8] = (f16)g0.x; a[9] = (f16)g0.y; a[10] = (f16)g0.z; a[11] = (f16)g0.w;
      a[12] = (f16)g1.x; a[13] = (f16)g1.y; a[14] = (f16)g1.z; a[15] = (f16)g1.w;
    }
#pragma unroll
    for (int nt = 0; nt < 4; ++nt) {
      v16h b = *(const v16h*)(sw1 + (k * 4 + nt) * 512 + lane * 16);
      acc[nt] = wmma_f16(a, b, acc[nt]);
    }
  }
#pragma unroll
  for (int nt = 0; nt < 4; ++nt) {
    const int n = nt * 16 + m;
    const float bias = b1[n];
#pragma unroll
    for (int r = 0; r < 8; ++r) {
      float v = acc[nt][r] + bias;
      v = v > 0.f ? v : 0.f;
      smid[wv][(r + hi * 8) * NH + n] = (f16)v;
    }
  }
  __syncthreads();

  // ---- GEMM2: mid [16x64] @ U2 [64x64] ----
  v8f acc2[4] = {};
#pragma unroll
  for (int k = 0; k < 2; ++k) {
    v8h c0 = *(const v8h*)(&smid[wv][m * NH + k * 32 + hi * 8]);
    v8h c1 = *(const v8h*)(&smid[wv][m * NH + k * 32 + 16 + hi * 8]);
    v16h a;
#pragma unroll
    for (int i = 0; i < 8; ++i) { a[i] = c0[i]; a[i + 8] = c1[i]; }
#pragma unroll
    for (int nt = 0; nt < 4; ++nt) {
      v16h b = *(const v16h*)(sw2 + (k * 4 + nt) * 512 + lane * 16);
      acc2[nt] = wmma_f16(a, b, acc2[nt]);
    }
  }
  // bias + relu, write u, accumulate per-graph moments
#pragma unroll
  for (int nt = 0; nt < 4; ++nt) {
    const int n = nt * 16 + m;
    const float bias = b2[n];
    float s = 0.f, sq = 0.f;
#pragma unroll
    for (int r = 0; r < 8; ++r) {
      float v = acc2[nt][r] + bias;
      v = v > 0.f ? v : 0.f;
      u[(tile * 16 + r + hi * 8) * NH + n] = v;
      s += v;
      sq += v * v;
    }
    s  += __shfl_xor(s, 16);
    sq += __shfl_xor(sq, 16);
    if (hi == 0) {
      unsafeAtomicAdd(&gsum[g * NH + n], s);
      unsafeAtomicAdd(&gsq[g * NH + n], sq);
    }
  }
}

// ---------------------------------------------------------------------------
// InstanceNorm: h = (u - mean) * rsqrt(var + eps), per graph (4096 nodes each)
// ---------------------------------------------------------------------------
__global__ void norm_kernel(const float* __restrict__ u,
                            const float* __restrict__ gsum,
                            const float* __restrict__ gsq,
                            float* __restrict__ hf, f16* __restrict__ hh) {
  int idx = blockIdx.x * blockDim.x + threadIdx.x;  // node*64 + f
  int f = idx & 63, node = idx >> 6, g = node >> 12;
  const float inv = 1.f / 4096.f;
  float mean = gsum[g * NH + f] * inv;
  float var  = gsq[g * NH + f] * inv - mean * mean;
  float v = (u[idx] - mean) * rsqrtf(var + EPSN);
  hf[idx] = v;
  hh[idx] = (f16)v;
}

// ---------------------------------------------------------------------------
// Decoder: out = h @ dec_W + dec_b   (N=8 -> scalar f32)
// ---------------------------------------------------------------------------
__global__ void dec_kernel(const float* __restrict__ hf, const float* __restrict__ W,
                           const float* __restrict__ b, float* __restrict__ out) {
  int idx = blockIdx.x * blockDim.x + threadIdx.x;  // node*8 + d
  int d = idx & 7, node = idx >> 3;
  float acc = b[d];
#pragma unroll
  for (int k = 0; k < 64; ++k) acc += hf[node * NH + k] * W[k * 8 + d];
  out[idx] = acc;
}

// ---------------------------------------------------------------------------
extern "C" void kernel_launch(void* const* d_in, const int* in_sizes, int n_in,
                              void* d_out, int out_size, void* d_ws, size_t ws_size,
                              hipStream_t stream) {
  const float* x    = (const float*)d_in[0];
  const int*   ei   = (const int*)d_in[1];
  // d_in[2] = batch (structure is fixed: 8 graphs x 4096 sorted nodes, unused)
  const float* encW = (const float*)d_in[3];
  const float* encB = (const float*)d_in[4];
  const float* decW = (const float*)d_in[5];
  const float* decB = (const float*)d_in[6];
  const float* m1W  = (const float*)d_in[7];
  const float* m1B  = (const float*)d_in[8];
  const float* m2W  = (const float*)d_in[9];
  const float* m2B  = (const float*)d_in[10];
  const float* u1W  = (const float*)d_in[11];
  const float* u1B  = (const float*)d_in[12];
  const float* u2W  = (const float*)d_in[13];
  const float* u2B  = (const float*)d_in[14];
  float* out = (float*)d_out;

  // workspace layout (256B aligned chunks)
  char* ws = (char*)d_ws;
  float* hf   = (float*)ws;                 ws += (size_t)NN * NH * 4;   // 8 MB
  f16*   hh   = (f16*)ws;                   ws += (size_t)NN * NH * 2;   // 4 MB
  float* agg  = (float*)ws;                 ws += (size_t)NN * NH * 4;   // 8 MB
  float* u    = (float*)ws;                 ws += (size_t)NN * NH * 4;   // 8 MB
  float* gsum = (float*)ws;                 ws += 8 * NH * 4;
  float* gsq  = (float*)ws;                 ws += 8 * NH * 4;
  f16* w1pk   = (f16*)ws;                   ws += 3 * 128 * 64 * 2;
  f16* w2pk   = (f16*)ws;                   ws += 3 * 64 * 64 * 2;
  f16* u1pk   = (f16*)ws;                   ws += 3 * 128 * 64 * 2;
  f16* u2pk   = (f16*)ws;                   ws += 3 * 64 * 64 * 2;

  // pack weights into WMMA B-fragment layout (idempotent, every launch)
  for (int l = 0; l < 3; ++l) {
    pack_b_kernel<<<(128 * 64 + 255) / 256, 256, 0, stream>>>(
        m1W + l * 128 * 64, w1pk + l * 128 * 64, 128, 64);
    pack_b_kernel<<<(64 * 64 + 255) / 256, 256, 0, stream>>>(
        m2W + l * 64 * 64, w2pk + l * 64 * 64, 64, 64);
    pack_b_kernel<<<(128 * 64 + 255) / 256, 256, 0, stream>>>(
        u1W + l * 128 * 64, u1pk + l * 128 * 64, 128, 64);
    pack_b_kernel<<<(64 * 64 + 255) / 256, 256, 0, stream>>>(
        u2W + l * 64 * 64, u2pk + l * 64 * 64, 64, 64);
  }

  enc_kernel<<<NN * NH / 256, 256, 0, stream>>>(x, encW, encB, hf, hh);

  for (int l = 0; l < 3; ++l) {
    zero_kernel<<<NN * NH / 256, 256, 0, stream>>>(agg, gsum, gsq);
    edge_msg_kernel<<<NE / 16 / 8, 256, 0, stream>>>(
        hh, ei, w1pk + l * 128 * 64, w2pk + l * 64 * 64,
        m1B + l * 64, m2B + l * 64, agg);
    node_upd_kernel<<<NN / 16 / 8, 256, 0, stream>>>(
        hh, agg, u1pk + l * 128 * 64, u2pk + l * 64 * 64,
        u1B + l * 64, u2B + l * 64, u, gsum, gsq);
    norm_kernel<<<NN * NH / 256, 256, 0, stream>>>(u, gsum, gsq, hf, hh);
  }

  dec_kernel<<<NN * 8 / 256, 256, 0, stream>>>(hf, decW, decB, out);
}
